// GraphAttentionLayer_14319420965100
// MI455X (gfx1250) — compile-verified, matrix operand order checked by reading, not run on hardware
//
#include <hip/hip_runtime.h>
#include <hip/hip_bf16.h>

// GAT layer: B=8, N=2048, in_c=out_c=64.
//  Kernel1: h = inp@W via WMMA bf16 (f32 acc); emit e_i, e_j (f32) and hT (bf16, transposed).
//  Kernel2: per 16-row tile: stage adj slab (128KB) in LDS once (async-to-LDS), two-pass
//           softmax from LDS, h' = P @ h via v_wmma_f32_16x16x32_bf16, ds_add_f32 reduce.

typedef __attribute__((ext_vector_type(16))) __bf16 v16bf;
typedef __attribute__((ext_vector_type(8)))  __bf16 v8bf;
typedef __attribute__((ext_vector_type(8)))  float  v8f;
typedef __attribute__((ext_vector_type(4)))  int    v4i;

#define GAT_B   8
#define GAT_N   2048
#define GAT_C   64
#define NEG_BIG (-1.0e30f)
#define LEAKY_SLOPE 0.01f

#if defined(__gfx1250__) && __has_builtin(__builtin_amdgcn_global_load_async_to_lds_b128)
#define HAVE_ASYNC_LDS 1
typedef __attribute__((address_space(1))) v4i* gbl_v4i_ptr;
typedef __attribute__((address_space(3))) v4i* lds_v4i_ptr;
#else
#define HAVE_ASYNC_LDS 0
#endif

__device__ __forceinline__ float leaky(float x) {
    return x > 0.0f ? x : LEAKY_SLOPE * x;
}

// ---------------------------------------------------------------------------
// Kernel 1: one wave per 16-row tile of one batch.
//   h_tile(16x64,f32) = bf16(inp_tile 16x64) @ bf16(W 64x64)   [2 k-steps x 4 n-tiles]
//   -> LDS -> e_i/e_j (f32), hT (bf16 transposed: [b][c][n])
// ---------------------------------------------------------------------------
__global__ __launch_bounds__(32) void gat_h_proj_kernel(
    const float* __restrict__ inp,   // [B][N][64]
    const float* __restrict__ W,     // [64][64]
    const float* __restrict__ a,     // [128][1]  (a1 = a[0:64], a2 = a[64:128])
    unsigned short* __restrict__ hT_raw,  // [B][64][N] bf16
    float* __restrict__ eiG,         // [B][N]
    float* __restrict__ ejG)         // [B][N]
{
    __bf16* __restrict__ hT = (__bf16*)hT_raw;
    const int b   = blockIdx.y;
    const int i0  = blockIdx.x * 16;
    const int l   = threadIdx.x;       // 0..31
    const int lr  = l & 15;
    const int hi8 = (l >> 4) * 8;      // K-half offset per ISA 16-bit operand layout

    __shared__ float hLds[16 * GAT_C];

    // --- A operand (16x32 bf16) x 2 k-steps; ISA layout:
    //     elem idx<8  -> K = hi8 + idx ; idx>=8 -> K = 16 + hi8 + (idx-8)
    const float* arow = inp + ((size_t)(b * GAT_N + i0 + lr)) * GAT_C;
    v16bf A0, A1;
#pragma unroll
    for (int idx = 0; idx < 16; ++idx) {
        const int k = (idx < 8) ? (hi8 + idx) : (16 + hi8 + (idx - 8));
        A0[idx] = (__bf16)arow[k];
        A1[idx] = (__bf16)arow[32 + k];
    }

    v8f acc[4];
#pragma unroll
    for (int nt = 0; nt < 4; ++nt)
#pragma unroll
        for (int r = 0; r < 8; ++r) acc[nt][r] = 0.0f;

#pragma unroll
    for (int nt = 0; nt < 4; ++nt) {
        const int n = nt * 16 + lr;            // lane carries column N (B/C layout)
        v16bf B0, B1;
#pragma unroll
        for (int idx = 0; idx < 16; ++idx) {
            const int k = (idx < 8) ? (hi8 + idx) : (16 + hi8 + (idx - 8));
            B0[idx] = (__bf16)W[k * GAT_C + n];
            B1[idx] = (__bf16)W[(32 + k) * GAT_C + n];
        }
        acc[nt] = __builtin_amdgcn_wmma_f32_16x16x32_bf16(
            false, A0, false, B0, (short)0, acc[nt], false, false);
        acc[nt] = __builtin_amdgcn_wmma_f32_16x16x32_bf16(
            false, A1, false, B1, (short)0, acc[nt], false, false);
    }

    // Spill C (lane<16: M=r ; lane>=16: M=r+8 ; col = nt*16 + lane%16)
#pragma unroll
    for (int nt = 0; nt < 4; ++nt)
#pragma unroll
        for (int r = 0; r < 8; ++r)
            hLds[(r + hi8) * GAT_C + nt * 16 + lr] = acc[nt][r];
    __syncthreads();

    // hT (transposed bf16): lane handles columns l and l+32
#pragma unroll
    for (int cc = 0; cc < 2; ++cc) {
        const int c = l + cc * 32;
        __bf16* dst = hT + ((size_t)(b * GAT_C + c)) * GAT_N + i0;
#pragma unroll
        for (int i = 0; i < 16; ++i) dst[i] = (__bf16)hLds[i * GAT_C + c];
    }

    // e_i (lanes 0..15) / e_j (lanes 16..31), full f32 precision
    const float* av = a + ((l < 16) ? 0 : GAT_C);
    float s = 0.0f;
#pragma unroll
    for (int c = 0; c < GAT_C; ++c) s += hLds[lr * GAT_C + c] * av[c];
    if (l < 16) eiG[b * GAT_N + i0 + lr] = s;
    else        ejG[b * GAT_N + i0 + lr] = s;
}

// ---------------------------------------------------------------------------
// Kernel 2: 256 threads (8 waves) per 16-row tile.
//   LDS: adj slab 128KB + ej 8KB + h' accum 4KB + reduction scratch (~146KB -> 2 WG/WGP)
// ---------------------------------------------------------------------------
#define K2_SMEM_BYTES (16 * GAT_N * 4 /*adj*/ + GAT_N * 4 /*ej*/ + 16 * GAT_C * 4 /*hp*/ \
                       + 256 * 4 /*redM*/ + 256 * 4 /*redL*/ + 3 * 16 * 4 /*m,il,ei*/)

__global__ __launch_bounds__(256, 2) void gat_attn_kernel(
    const int* __restrict__ adj,            // [B][N][N]
    const unsigned short* __restrict__ hT_raw,  // [B][64][N] bf16
    const float* __restrict__ eiG,
    const float* __restrict__ ejG,
    float* __restrict__ out)                // [B][N][64]
{
    const __bf16* __restrict__ hT = (const __bf16*)hT_raw;
    const int b  = blockIdx.y;
    const int i0 = blockIdx.x * 16;
    const int t  = threadIdx.x;
    const int l  = t & 31;
    const int w  = t >> 5;

    extern __shared__ char smem[];
    int*   adjLds = (int*)smem;                                   // 16*2048
    float* ejLds  = (float*)(smem + 16 * GAT_N * 4);              // 2048
    float* hpLds  = ejLds + GAT_N;                                // 16*64
    float* redM   = hpLds + 16 * GAT_C;                           // 256
    float* redL   = redM + 256;                                   // 256
    float* mRow   = redL + 256;                                   // 16
    float* ilRow  = mRow + 16;                                    // 16
    float* eiRow  = ilRow + 16;                                   // 16

    // ---- stage: adj slab is one contiguous 128KB block (rows i0..i0+15) ----
    {
        int* adjGbase = const_cast<int*>(adj) + ((size_t)(b * GAT_N + i0)) * GAT_N;
        v4i* adjG = (v4i*)adjGbase;
        v4i* adjL = (v4i*)adjLds;
        for (int idx = t; idx < 16 * GAT_N / 4; idx += 256) {
#if HAVE_ASYNC_LDS
            __builtin_amdgcn_global_load_async_to_lds_b128(
                (gbl_v4i_ptr)(adjG + idx),
                (lds_v4i_ptr)(adjL + idx), 0, 0);
#else
            adjL[idx] = adjG[idx];
#endif
        }
        for (int idx = t; idx < GAT_N; idx += 256) ejLds[idx] = ejG[b * GAT_N + idx];
        if (t < 16) eiRow[t] = eiG[b * GAT_N + i0 + t];
        for (int idx = t; idx < 16 * GAT_C; idx += 256) hpLds[idx] = 0.0f;
    }
#if HAVE_ASYNC_LDS && __has_builtin(__builtin_amdgcn_s_wait_asynccnt)
    __builtin_amdgcn_s_wait_asynccnt(0);
#endif
    __syncthreads();

    // ---- pass A: per-row online max / sum-exp (16 threads per row) ----
    {
        const int row = t >> 4, sub = t & 15;
        const float eirow = eiRow[row];
        const int* arow = adjLds + row * GAT_N;
        float m = NEG_BIG, lsum = 0.0f;
        for (int j = sub; j < GAT_N; j += 16) {
            if (arow[j] > 0) {
                const float e = leaky(eirow + ejLds[j]);
                if (e > m) { lsum = lsum * __expf(m - e) + 1.0f; m = e; }
                else       { lsum += __expf(e - m); }
            }
        }
        redM[t] = m; redL[t] = lsum;
        __syncthreads();
        for (int s = 8; s > 0; s >>= 1) {
            if (sub < s) {
                const float m1 = redM[t], l1 = redL[t];
                const float m2 = redM[t + s], l2 = redL[t + s];
                const float mn = fmaxf(m1, m2);
                redM[t] = mn;
                redL[t] = l1 * __expf(m1 - mn) + l2 * __expf(m2 - mn);
            }
            __syncthreads();
        }
        if (sub == 0) {
            const float lf = redL[t];
            mRow[row]  = redM[t];
            ilRow[row] = lf > 0.0f ? 1.0f / lf : 0.0f;
        }
    }
    __syncthreads();

    // ---- pass B: h' = P @ h via WMMA; each wave takes column tiles w, w+8, ... ----
    const int lr  = l & 15;
    const int hi8 = (l >> 4) * 8;
    const float m_r  = mRow[lr];
    const float il_r = ilRow[lr];
    const float ei_r = eiRow[lr];
    const int* arow = adjLds + lr * GAT_N;

    v8f acc[4];
#pragma unroll
    for (int nt = 0; nt < 4; ++nt)
#pragma unroll
        for (int r = 0; r < 8; ++r) acc[nt][r] = 0.0f;

    for (int jt = w; jt < GAT_N / 32; jt += 8) {
        const int j0 = jt * 32;
        // A = probability tile (16x32 bf16), ISA A-layout
        v16bf Av;
#pragma unroll
        for (int idx = 0; idx < 16; ++idx) {
            const int k = (idx < 8) ? (hi8 + idx) : (16 + hi8 + (idx - 8));
            const int j = j0 + k;
            float p = 0.0f;
            if (arow[j] > 0) {
                const float e = leaky(ei_r + ejLds[j]);
                p = __expf(e - m_r) * il_r;
            }
            Av[idx] = (__bf16)p;
        }
        // B = hT rows (contiguous 16B chunks per lane, ISA B-layout)
#pragma unroll
        for (int nt = 0; nt < 4; ++nt) {
            const __bf16* hp = hT + ((size_t)(b * GAT_C + nt * 16 + lr)) * GAT_N + j0 + hi8;
            const v8bf c0 = *(const v8bf*)hp;          // K = hi8 .. hi8+7
            const v8bf c1 = *(const v8bf*)(hp + 16);   // K = 16+hi8 .. 16+hi8+7
            v16bf Bv;
#pragma unroll
            for (int i = 0; i < 8; ++i) { Bv[i] = c0[i]; Bv[8 + i] = c1[i]; }
            acc[nt] = __builtin_amdgcn_wmma_f32_16x16x32_bf16(
                false, Av, false, Bv, (short)0, acc[nt], false, false);
        }
    }

    // cross-wave reduce partial h' via ds_add_f32
#pragma unroll
    for (int nt = 0; nt < 4; ++nt)
#pragma unroll
        for (int r = 0; r < 8; ++r)
            atomicAdd(&hpLds[(r + hi8) * GAT_C + nt * 16 + lr], acc[nt][r]);
    __syncthreads();

    for (int idx = t; idx < 16 * GAT_C; idx += 256) {
        const int row = idx >> 6, c = idx & 63;
        out[((size_t)(b * GAT_N + i0 + row)) * GAT_C + c] = hpLds[idx];
    }
}

// ---------------------------------------------------------------------------
extern "C" void kernel_launch(void* const* d_in, const int* in_sizes, int n_in,
                              void* d_out, int out_size, void* d_ws, size_t ws_size,
                              hipStream_t stream) {
    const float* inp = (const float*)d_in[0];   // [8][2048][64] f32
    const int*   adj = (const int*)d_in[1];     // [8][2048][2048] i32
    const float* W   = (const float*)d_in[2];   // [64][64] f32
    const float* a   = (const float*)d_in[3];   // [128][1] f32
    float* out = (float*)d_out;                 // [8][2048][64] f32

    // workspace: hT bf16 (2 MB) | ei (64 KB) | ej (64 KB)
    char* ws = (char*)d_ws;
    unsigned short* hT = (unsigned short*)ws;
    float* eiG = (float*)(ws + (size_t)GAT_B * GAT_C * GAT_N * 2);
    float* ejG = eiG + GAT_B * GAT_N;

    dim3 grid(GAT_N / 16, GAT_B);
    gat_h_proj_kernel<<<grid, 32, 0, stream>>>(inp, W, a, hT, eiG, ejG);
    gat_attn_kernel<<<grid, 256, K2_SMEM_BYTES, stream>>>(adj, hT, eiG, ejG, out);
}
